// ScaledDotProductAttention_43190191128742
// MI455X (gfx1250) — matrix operand order repaired and checked by reading, not compiled, not run on hardware
//
#include <hip/hip_runtime.h>
#include <hip/hip_bf16.h>

// ---------------------------------------------------------------------------
// Causal scaled-dot-product attention forward for MI455X (gfx1250, wave32).
// Shapes fixed by the reference: B=2, H=16, S=2048, D=64, fp32 in/out.
// Output buffer = out[B,H,S,D] followed by attn[B,H,S,S] (tuple return order).
//
// Design: attn write (537 MB) dominates -> single fused pass, score strip for
// a 16-row query tile kept resident in LDS (16 x 2048 fp32 = 128 KB, padded),
// both GEMMs on v_wmma_f32_16x16x32_f16.
// ---------------------------------------------------------------------------

typedef __attribute__((ext_vector_type(16))) _Float16 v16h;
typedef __attribute__((ext_vector_type(8)))  float    v8f;

#define B_  2
#define H_  16
#define S_  2048
#define D_  64
#define QT  16                 // query rows per workgroup
#define LDS_STRIDE (S_ + 4)    // +4 floats -> rows land on distinct LDS banks

__device__ __forceinline__ float wave_max32(float v) {
#pragma unroll
  for (int off = 16; off >= 1; off >>= 1)
    v = fmaxf(v, __shfl_xor(v, off, 32));
  return v;
}
__device__ __forceinline__ float wave_sum32(float v) {
#pragma unroll
  for (int off = 16; off >= 1; off >>= 1)
    v += __shfl_xor(v, off, 32);
  return v;
}

__global__ __launch_bounds__(256)
void sdpa_fwd_wmma_kernel(const float* __restrict__ q,
                          const float* __restrict__ k,
                          const float* __restrict__ v,
                          float* __restrict__ out,
                          float* __restrict__ attn) {
  // 16-row score/probability strip, fp32, padded stride (131,328 B of LDS).
  __shared__ float sP[QT * LDS_STRIDE];
  __shared__ float invsum_sh[QT];

  const int qt   = blockIdx.x;           // query tile index (0..127)
  const int h    = blockIdx.y;
  const int b    = blockIdx.z;
  const int tid  = threadIdx.x;
  const int wave = tid >> 5;             // 8 wave32's
  const int lane = tid & 31;
  const int lo16 = lane & 15;
  const int half = lane >> 4;            // lane half (WMMA layout split)

  const int    q0   = qt * QT;
  const int    kend = (qt + 1) * QT;     // causal: keys [0, kend) matter
  const size_t bh   = (size_t)b * H_ + h;
  const float* qbase = q + bh * S_ * D_;
  const float* kbase = k + bh * S_ * D_;
  const float* vbase = v + bh * S_ * D_;

  // -------- Build Q A-fragments once (M=16, K=64 split into two K=32 chunks).
  // A k order used here (contiguous, same as B): lane m=lane&15; element i -> K = 16*half + i.
  v16h aq0, aq1;
  {
    const float* qrow = qbase + (size_t)(q0 + lo16) * D_;
#pragma unroll
    for (int i = 0; i < 16; ++i) {
      const int j = i >> 1, sub = i & 1;
      const int kk = (half << 4) + i;
      aq0[i] = (_Float16)qrow[kk];
      aq1[i] = (_Float16)qrow[32 + kk];
    }
  }

  // -------- Phase 1: score tiles. Wave w handles key tiles w, w+8, ...
  // scores = (Q . K^T) / 8, diagonal tile masked to -1e5 (matches reference).
  for (int kt = wave; kt <= qt; kt += 8) {
    // B k order used here (contiguous, same as A): lane n=lane&15; element i -> K = 16*half + i.
    // B[k][n] = Ktile[n][k]  => contiguous 16-float gather per lane.
    const float* krow = kbase + (size_t)(kt * QT + lo16) * D_ + (half << 4);
    v16h bk0, bk1;
#pragma unroll
    for (int i = 0; i < 16; ++i) {
      bk0[i] = (_Float16)krow[i];
      bk1[i] = (_Float16)krow[32 + i];
    }
    v8f c = {};
    c = __builtin_amdgcn_wmma_f32_16x16x32_f16(false, aq0, false, bk0,
                                               (short)0, c, false, false);
    c = __builtin_amdgcn_wmma_f32_16x16x32_f16(false, aq1, false, bk1,
                                               (short)0, c, false, false);
    // C/D layout: element i -> row m = i + 8*half, col n = lane&15.
#pragma unroll
    for (int i = 0; i < 8; ++i) {
      const int row = i + (half << 3);
      float s = c[i] * 0.125f;                       // 1/sqrt(64)
      if (kt == qt && lo16 > row) s = -100000.0f;    // causal mask on diagonal
      sP[row * LDS_STRIDE + kt * QT + lo16] = s;
    }
  }
  __syncthreads();

  // -------- Phase 2: softmax. Wave w owns rows 2w, 2w+1 (lane-parallel cols).
#pragma unroll
  for (int rr = 0; rr < 2; ++rr) {
    const int r = wave * 2 + rr;
    float* prow = sP + r * LDS_STRIDE;
    float mx = -3.0e38f;
    for (int c = lane; c < kend; c += 32) mx = fmaxf(mx, prow[c]);
    mx = wave_max32(mx);
    float sum = 0.0f;
    for (int c = lane; c < kend; c += 32) {
      const float p = __expf(prow[c] - mx);          // masked -> exactly 0
      prow[c] = p;                                   // store unnormalized p
      sum += p;
    }
    sum = wave_sum32(sum);
    if (lane == 0) invsum_sh[r] = 1.0f / sum;
  }
  __syncthreads();

  if (wave < 4) {
    // ---- Phase 3a (waves 0-3): out[16 x 64] = P @ V. Wave w owns D-columns
    // [16w, 16w+16); K dim = key positions, chunked by 32 (2 key tiles).
    const int nt = wave;
    v8f acc = {};
    const int nchunks = (qt + 2) >> 1;               // ceil(kend / 32)
    for (int c = 0; c < nchunks; ++c) {
      const int kb = c * 32;
      v16h ap, bv;
#pragma unroll
      for (int i = 0; i < 16; ++i) {                 // A frag from LDS p
        const int j = i >> 1, sub = i & 1;
        const int kl  = (half << 4) + i;
        const int col = kb + kl;
        ap[i] = (_Float16)((col < kend) ? sP[lo16 * LDS_STRIDE + col] : 0.0f);
      }
#pragma unroll
      for (int i = 0; i < 16; ++i) {                 // B frag from V (global)
        const int krow = kb + (half << 4) + i;
        const float vv = (krow < kend)
            ? vbase[(size_t)krow * D_ + nt * QT + lo16] : 0.0f;
        bv[i] = (_Float16)vv;
      }
      acc = __builtin_amdgcn_wmma_f32_16x16x32_f16(false, ap, false, bv,
                                                   (short)0, acc, false, false);
    }
#pragma unroll
    for (int i = 0; i < 8; ++i) {                    // normalize at store
      const int m = i + (half << 3);
      out[(bh * S_ + q0 + m) * D_ + nt * QT + lo16] = acc[i] * invsum_sh[m];
    }
  } else {
    // ---- Phase 3b (waves 4-7): stream normalized attn rows + causal zeros.
    const int t = tid - 128;                         // 0..127
    float* arow0 = attn + (bh * S_ + q0) * (size_t)S_;
    for (int idx = t; idx < QT * (S_ / 4); idx += 128) {
      const int r     = idx >> 9;                    // S_/4 == 512
      const int c4    = (idx & 511) << 2;
      const int limit = q0 + r + 1;                  // keys <= query index
      const float is  = invsum_sh[r];
      const float* prow = sP + r * LDS_STRIDE;
      float4 val;
      val.x = (c4 + 0 < limit) ? prow[c4 + 0] * is : 0.0f;
      val.y = (c4 + 1 < limit) ? prow[c4 + 1] * is : 0.0f;
      val.z = (c4 + 2 < limit) ? prow[c4 + 2] * is : 0.0f;
      val.w = (c4 + 3 < limit) ? prow[c4 + 3] * is : 0.0f;
      *reinterpret_cast<float4*>(arow0 + (size_t)r * S_ + c4) = val;
    }
  }
}

extern "C" void kernel_launch(void* const* d_in, const int* in_sizes, int n_in,
                              void* d_out, int out_size, void* d_ws, size_t ws_size,
                              hipStream_t stream) {
  (void)in_sizes; (void)n_in; (void)out_size; (void)d_ws; (void)ws_size;
  const float* q = (const float*)d_in[0];
  const float* k = (const float*)d_in[1];
  const float* v = (const float*)d_in[2];
  // d_in[3] (mask) unused: causality is applied analytically in-kernel.
  float* out  = (float*)d_out;
  float* attn = out + (size_t)B_ * H_ * S_ * D_;     // tuple order: (out, attn)

  dim3 grid(S_ / QT, H_, B_);                        // 128 x 16 x 2
  dim3 block(256);                                   // 8 wave32 per WG
  sdpa_fwd_wmma_kernel<<<grid, block, 0, stream>>>(q, k, v, out, attn);
}
